// SlotCausalAttention_18863496364353
// MI455X (gfx1250) — matrix operand order repaired and checked by reading, not compiled, not run on hardware
//
#include <hip/hip_runtime.h>

// ---------------------------------------------------------------------------
// SlotCausalAttention for MI455X (gfx1250), wave32 + WMMA bf16 16x16x32.
// Pipeline: [bf16 pre-convert] -> [qkv GEMM] -> [flash attention] -> [out GEMM]
// GEMM staging uses GLOBAL_LOAD_ASYNC_TO_LDS_B128 (ASYNCcnt) when available.
// ---------------------------------------------------------------------------

#define B_ 4
#define N_ 1024
#define M_ 256
#define L_ 1280          // N_ + M_
#define E_ 1024
#define H_ 16
#define D_ 64            // E_ / H_

typedef __attribute__((ext_vector_type(16))) __bf16 v16bf;
typedef __attribute__((ext_vector_type(8)))  float  v8f;

union BF16Frag {
    v16bf v;
    uint4 q[2];
};

static __device__ __forceinline__ unsigned short f2bf(float f) {
    unsigned int u = __float_as_uint(f);
    u += 0x7FFFu + ((u >> 16) & 1u);   // round-to-nearest-even
    return (unsigned short)(u >> 16);
}

static __device__ __forceinline__ v8f wmma_bf16(const BF16Frag& a, const BF16Frag& b, v8f c) {
    return __builtin_amdgcn_wmma_f32_16x16x32_bf16(
        /*neg_a=*/false, a.v, /*neg_b=*/false, b.v,
        /*c_mod=*/(short)0, c, /*reuse_a=*/false, /*reuse_b=*/false);
}

// ---- async global->LDS copy (gfx1250 GLOBAL_LOAD_ASYNC_TO_LDS_B128) --------
#if __has_builtin(__builtin_amdgcn_global_load_async_to_lds_b128) && \
    __has_builtin(__builtin_amdgcn_s_wait_asynccnt)
typedef int v4i __attribute__((vector_size(16)));            // GCC-style vector per builtin sig
typedef __attribute__((address_space(1))) v4i gv4i;          // global AS ("__device__")
typedef __attribute__((address_space(3))) v4i lv4i;          // LDS AS
static __device__ __forceinline__ void async_copy16(void* lds_dst, const void* gsrc) {
    __builtin_amdgcn_global_load_async_to_lds_b128((gv4i*)gsrc, (lv4i*)lds_dst, 0, 0);
}
static __device__ __forceinline__ void async_wait() {
    __builtin_amdgcn_s_wait_asynccnt(0);
}
#else
static __device__ __forceinline__ void async_copy16(void* lds_dst, const void* gsrc) {
    *(uint4*)lds_dst = *(const uint4*)gsrc;
}
static __device__ __forceinline__ void async_wait() {}
#endif

// ---------------------------------------------------------------------------
// Kernel 0a: build concatenated bf16 activations xs[B][L][E] from x and slots.
// Block = 256 threads handles 2 rows; grid.x = B*L/2.
// ---------------------------------------------------------------------------
__global__ __launch_bounds__(256) void build_xs_kernel(
        const float* __restrict__ x, const float* __restrict__ slots,
        unsigned short* __restrict__ xs) {
    const int rb  = blockIdx.x * 2 + (threadIdx.x >> 7);
    const int col = (threadIdx.x & 127) * 8;
    const int b = rb / L_;
    const int l = rb % L_;
    const float* src = (l < N_) ? x + ((size_t)b * N_ + l) * E_ + col
                                : slots + ((size_t)b * M_ + (l - N_)) * E_ + col;
    float4 a = *(const float4*)src;
    float4 c = *(const float4*)(src + 4);
    uint4 o;
    o.x = f2bf(a.x) | ((unsigned)f2bf(a.y) << 16);
    o.y = f2bf(a.z) | ((unsigned)f2bf(a.w) << 16);
    o.z = f2bf(c.x) | ((unsigned)f2bf(c.y) << 16);
    o.w = f2bf(c.z) | ((unsigned)f2bf(c.w) << 16);
    *(uint4*)(xs + (size_t)rb * E_ + col) = o;
}

// Kernel 0b: elementwise f32 -> bf16 (exact multiple of 2048 elems per grid).
__global__ __launch_bounds__(256) void cvt_bf16_kernel(
        const float* __restrict__ src, unsigned short* __restrict__ dst) {
    const size_t i = ((size_t)blockIdx.x * 256 + threadIdx.x) * 8;
    float4 a = *(const float4*)(src + i);
    float4 c = *(const float4*)(src + i + 4);
    uint4 o;
    o.x = f2bf(a.x) | ((unsigned)f2bf(a.y) << 16);
    o.y = f2bf(a.z) | ((unsigned)f2bf(a.w) << 16);
    o.z = f2bf(c.x) | ((unsigned)f2bf(c.y) << 16);
    o.w = f2bf(c.z) | ((unsigned)f2bf(c.w) << 16);
    *(uint4*)(dst + i) = o;
}

// ---------------------------------------------------------------------------
// Kernel 1: proj = xs @ w_in^T + b_in ; split into Q*scale, K, V (bf16 BHLD).
// Grid: (3E/64, L/256, B), block 256 (8 waves).
// Wave tile: 32(M) x 64(N) = 8 WMMAs per 32-deep k-step.
// ---------------------------------------------------------------------------
__global__ __launch_bounds__(256) void qkv_proj_kernel(
        const unsigned short* __restrict__ xs, const unsigned short* __restrict__ wi,
        const float* __restrict__ b_in,
        unsigned short* __restrict__ Qb, unsigned short* __restrict__ Kb,
        unsigned short* __restrict__ Vb) {
    __shared__ unsigned short sA[256][32];   // 16 KB, [row][k] bf16
    __shared__ unsigned short sB[64][32];    //  4 KB, [n][k]  bf16

    const int t      = threadIdx.x;
    const int lane   = t & 31;
    const int wave   = t >> 5;
    const int lane15 = lane & 15;
    const int khalf  = lane >> 4;
    const int b      = blockIdx.z;
    const int tm     = blockIdx.y;           // rows [tm*256, tm*256+256)
    const int col0   = blockIdx.x * 64;

    const unsigned short* aptr = xs + ((size_t)b * L_ + tm * 256 + t) * E_;  // thread t stages row t
    const unsigned short* bptr = wi + (size_t)(col0 + (t >> 2)) * E_ + (t & 3) * 8;

    v8f acc[2][4] = {};

    for (int k0 = 0; k0 < E_; k0 += 32) {
        // ---- async stage: A 256x32 bf16 (4x16B per thread), B 64x32 (1x16B) ----
#pragma unroll
        for (int i = 0; i < 4; ++i) async_copy16(&sA[t][i * 8], aptr + k0 + i * 8);
        async_copy16(&sB[t >> 2][(t & 3) * 8], bptr + k0);
        async_wait();
        __syncthreads();

        BF16Frag bfr[4];
#pragma unroll
        for (int s = 0; s < 4; ++s) {
            bfr[s].q[0] = *(const uint4*)&sB[s * 16 + lane15][khalf * 8];
            bfr[s].q[1] = *(const uint4*)&sB[s * 16 + lane15][16 + khalf * 8];
        }
#pragma unroll
        for (int mi = 0; mi < 2; ++mi) {
            const int mrow = wave * 32 + mi * 16 + lane15;
            BF16Frag afr;
            afr.q[0] = *(const uint4*)&sA[mrow][khalf * 8];
            afr.q[1] = *(const uint4*)&sA[mrow][16 + khalf * 8];
#pragma unroll
            for (int s = 0; s < 4; ++s) acc[mi][s] = wmma_bf16(afr, bfr[s], acc[mi][s]);
        }
        __syncthreads();
    }

    // ---- epilogue: bias, q-scale, scatter to [B][H][L][D] bf16 ----
#pragma unroll
    for (int mi = 0; mi < 2; ++mi) {
        const int lrow0 = tm * 256 + wave * 32 + mi * 16 + khalf * 8;
#pragma unroll
        for (int s = 0; s < 4; ++s) {
            const int col  = col0 + s * 16 + lane15;
            const float bias = b_in[col];
            const int sel = col / E_;        // 0=q 1=k 2=v
            const int ce  = col % E_;
            const int h   = ce / D_;
            const int dim = ce % D_;
            unsigned short* dst = (sel == 0) ? Qb : ((sel == 1) ? Kb : Vb);
            const float scale = (sel == 0) ? 0.125f : 1.0f;   // 1/sqrt(64)
#pragma unroll
            for (int j = 0; j < 8; ++j) {
                const int lr = lrow0 + j;
                dst[(((size_t)b * H_ + h) * L_ + lr) * D_ + dim] =
                    f2bf((acc[mi][s][j] + bias) * scale);
            }
        }
    }
}

// ---------------------------------------------------------------------------
// Kernel 2: flash attention with additive 0/1 mask bias (torch float-mask add).
// Grid: (L/128, H, B), block 256 (8 waves); wave owns 16 query rows.
// K tile staged via async-to-LDS; V tile transposed manually.
// ---------------------------------------------------------------------------
__global__ __launch_bounds__(256) void attn_kernel(
        const unsigned short* __restrict__ Qb, const unsigned short* __restrict__ Kb,
        const unsigned short* __restrict__ Vb, unsigned short* __restrict__ attnb) {
    __shared__ unsigned short sK[32][64];       // 4 KB, [key][dim]
    __shared__ unsigned short sV[64][32];       // 4 KB, transposed [dim][key]
    __shared__ unsigned short sP[8][16][32];    // 8 KB, per-wave P scratch

    const int t      = threadIdx.x;
    const int lane   = t & 31;
    const int wave   = t >> 5;
    const int lane15 = lane & 15;
    const int khalf  = lane >> 4;
    const int b      = blockIdx.z;
    const int h      = blockIdx.y;
    const int qrow0  = blockIdx.x * 128 + wave * 16;

    const unsigned short* Qbase = Qb + (((size_t)b * H_ + h) * L_) * D_;
    const unsigned short* Kbase = Kb + (((size_t)b * H_ + h) * L_) * D_;
    const unsigned short* Vbase = Vb + (((size_t)b * H_ + h) * L_) * D_;

    // Q fragments (reused across whole key loop): 2 k-steps of 32 features
    BF16Frag qa[2];
#pragma unroll
    for (int s = 0; s < 2; ++s) {
        const unsigned short* qp = Qbase + (size_t)(qrow0 + lane15) * D_ + s * 32;
        qa[s].q[0] = *(const uint4*)(qp + khalf * 8);
        qa[s].q[1] = *(const uint4*)(qp + 16 + khalf * 8);
    }

    v8f   acc[4] = {};
    float mj[8], lsum[8];
#pragma unroll
    for (int j = 0; j < 8; ++j) { mj[j] = -1e30f; lsum[j] = 0.0f; }

    const int key = t >> 3;              // 0..31
    const int dco = (t & 7) * 8;         // 0..56

    for (int kb = 0; kb < L_; kb += 32) {
        // ---- stage K via async copy; V with manual transpose ----
        async_copy16(&sK[key][dco], Kbase + (size_t)(kb + key) * D_ + dco);
        uint4 vv = *(const uint4*)(Vbase + (size_t)(kb + key) * D_ + dco);
        sV[dco + 0][key] = (unsigned short)(vv.x);
        sV[dco + 1][key] = (unsigned short)(vv.x >> 16);
        sV[dco + 2][key] = (unsigned short)(vv.y);
        sV[dco + 3][key] = (unsigned short)(vv.y >> 16);
        sV[dco + 4][key] = (unsigned short)(vv.z);
        sV[dco + 5][key] = (unsigned short)(vv.z >> 16);
        sV[dco + 6][key] = (unsigned short)(vv.w);
        sV[dco + 7][key] = (unsigned short)(vv.w >> 16);
        async_wait();
        __syncthreads();

        // ---- scores: S[16 q x 32 keys] = Q(16x64) . K^T ----
        v8f s0 = {}, s1 = {};
#pragma unroll
        for (int s = 0; s < 2; ++s) {          // feature halves (GEMM K-dim)
            BF16Frag kf0, kf1;
            kf0.q[0] = *(const uint4*)&sK[lane15][s * 32 + khalf * 8];
            kf0.q[1] = *(const uint4*)&sK[lane15][s * 32 + 16 + khalf * 8];
            kf1.q[0] = *(const uint4*)&sK[16 + lane15][s * 32 + khalf * 8];
            kf1.q[1] = *(const uint4*)&sK[16 + lane15][s * 32 + 16 + khalf * 8];
            s0 = wmma_bf16(qa[s], kf0, s0);
            s1 = wmma_bf16(qa[s], kf1, s1);
        }

        // ---- online softmax (additive 0/1 bias; NOT -inf masking) ----
        const int colk0 = kb + lane15;
        const int colk1 = colk0 + 16;
#pragma unroll
        for (int j = 0; j < 8; ++j) {
            const int qrow = qrow0 + j + 8 * khalf;
            float v0 = s0[j] + ((colk0 < N_ || qrow >= colk0) ? 1.0f : 0.0f);
            float v1 = s1[j] + ((colk1 < N_ || qrow >= colk1) ? 1.0f : 0.0f);
            float mx = fmaxf(v0, v1);
#pragma unroll
            for (int off = 1; off < 16; off <<= 1)
                mx = fmaxf(mx, __shfl_xor(mx, off, 32));     // rows live in 16-lane halves
            const float newm = fmaxf(mj[j], mx);
            const float corr = __expf(mj[j] - newm);
            mj[j] = newm;
            const float p0 = __expf(v0 - newm);
            const float p1 = __expf(v1 - newm);
            float ps = p0 + p1;
#pragma unroll
            for (int off = 1; off < 16; off <<= 1)
                ps += __shfl_xor(ps, off, 32);
            lsum[j] = lsum[j] * corr + ps;
#pragma unroll
            for (int s2 = 0; s2 < 4; ++s2) acc[s2][j] *= corr;
            sP[wave][j + 8 * khalf][lane15]      = f2bf(p0);
            sP[wave][j + 8 * khalf][16 + lane15] = f2bf(p1);
        }

        // ---- PV: A = P(16x32), B = V^T tiles, 4 WMMAs into acc ----
        BF16Frag pa;
        pa.q[0] = *(const uint4*)&sP[wave][lane15][khalf * 8];
        pa.q[1] = *(const uint4*)&sP[wave][lane15][16 + khalf * 8];
#pragma unroll
        for (int s2 = 0; s2 < 4; ++s2) {
            BF16Frag vf;
            vf.q[0] = *(const uint4*)&sV[s2 * 16 + lane15][khalf * 8];
            vf.q[1] = *(const uint4*)&sV[s2 * 16 + lane15][16 + khalf * 8];
            acc[s2] = wmma_bf16(pa, vf, acc[s2]);
        }
        __syncthreads();
    }

    // ---- normalize and write bf16 [B][L][E] ----
#pragma unroll
    for (int j = 0; j < 8; ++j) {
        const float inv = 1.0f / lsum[j];
        const int lr = qrow0 + j + 8 * khalf;
#pragma unroll
        for (int s2 = 0; s2 < 4; ++s2) {
            attnb[((size_t)b * L_ + lr) * E_ + h * D_ + s2 * 16 + lane15] =
                f2bf(acc[s2][j] * inv);
        }
    }
}

// ---------------------------------------------------------------------------
// Kernel 3: out = attn @ w_out^T + b_out, written as (out[:, :n], out[:, n:])
// concatenated flat.  Grid: (E/64, L/256, B), block 256.
// ---------------------------------------------------------------------------
__global__ __launch_bounds__(256) void out_proj_kernel(
        const unsigned short* __restrict__ attnb, const unsigned short* __restrict__ wo,
        const float* __restrict__ b_out, float* __restrict__ out) {
    __shared__ unsigned short sA[256][32];
    __shared__ unsigned short sB[64][32];

    const int t      = threadIdx.x;
    const int lane   = t & 31;
    const int wave   = t >> 5;
    const int lane15 = lane & 15;
    const int khalf  = lane >> 4;
    const int b      = blockIdx.z;
    const int tm     = blockIdx.y;
    const int col0   = blockIdx.x * 64;

    const unsigned short* aptr = attnb + ((size_t)b * L_ + tm * 256 + t) * E_;
    const unsigned short* bptr = wo + (size_t)(col0 + (t >> 2)) * E_ + (t & 3) * 8;

    v8f acc[2][4] = {};

    for (int k0 = 0; k0 < E_; k0 += 32) {
#pragma unroll
        for (int i = 0; i < 4; ++i) async_copy16(&sA[t][i * 8], aptr + k0 + i * 8);
        async_copy16(&sB[t >> 2][(t & 3) * 8], bptr + k0);
        async_wait();
        __syncthreads();

        BF16Frag bfr[4];
#pragma unroll
        for (int s = 0; s < 4; ++s) {
            bfr[s].q[0] = *(const uint4*)&sB[s * 16 + lane15][khalf * 8];
            bfr[s].q[1] = *(const uint4*)&sB[s * 16 + lane15][16 + khalf * 8];
        }
#pragma unroll
        for (int mi = 0; mi < 2; ++mi) {
            const int mrow = wave * 32 + mi * 16 + lane15;
            BF16Frag afr;
            afr.q[0] = *(const uint4*)&sA[mrow][khalf * 8];
            afr.q[1] = *(const uint4*)&sA[mrow][16 + khalf * 8];
#pragma unroll
            for (int s = 0; s < 4; ++s) acc[mi][s] = wmma_bf16(afr, bfr[s], acc[mi][s]);
        }
        __syncthreads();
    }

#pragma unroll
    for (int mi = 0; mi < 2; ++mi) {
        const int lrow0 = tm * 256 + wave * 32 + mi * 16 + khalf * 8;
#pragma unroll
        for (int s = 0; s < 4; ++s) {
            const int col  = col0 + s * 16 + lane15;
            const float bias = b_out[col];
#pragma unroll
            for (int j = 0; j < 8; ++j) {
                const int lr = lrow0 + j;
                const float val = acc[mi][s][j] + bias;
                size_t off;
                if (lr < N_) off = ((size_t)b * N_ + lr) * E_ + col;
                else         off = (size_t)B_ * N_ * E_ + ((size_t)b * M_ + (lr - N_)) * E_ + col;
                out[off] = val;
            }
        }
    }
}

// ---------------------------------------------------------------------------
extern "C" void kernel_launch(void* const* d_in, const int* in_sizes, int n_in,
                              void* d_out, int out_size, void* d_ws, size_t ws_size,
                              hipStream_t stream) {
    const float* x     = (const float*)d_in[0];
    const float* slots = (const float*)d_in[1];
    const float* w_in  = (const float*)d_in[2];
    const float* b_in  = (const float*)d_in[3];
    const float* w_out = (const float*)d_in[4];
    const float* b_out = (const float*)d_in[5];
    float* out = (float*)d_out;

    // workspace layout (all bf16/ushort):
    //   xs [B,L,E] | wi [3E,E] | wo [E,E] | Q,K,V [B,H,L,D] | attn [B,L,E]
    unsigned short* xs = (unsigned short*)d_ws;
    unsigned short* wi = xs + (size_t)B_ * L_ * E_;
    unsigned short* wo = wi + (size_t)3 * E_ * E_;
    unsigned short* Qb = wo + (size_t)E_ * E_;
    unsigned short* Kb = Qb + (size_t)B_ * H_ * L_ * D_;
    unsigned short* Vb = Kb + (size_t)B_ * H_ * L_ * D_;
    unsigned short* attnb = Vb + (size_t)B_ * H_ * L_ * D_;

    build_xs_kernel<<<dim3(B_ * L_ / 2), 256, 0, stream>>>(x, slots, xs);
    cvt_bf16_kernel<<<dim3(3 * E_ * E_ / 2048), 256, 0, stream>>>(w_in, wi);
    cvt_bf16_kernel<<<dim3(E_ * E_ / 2048), 256, 0, stream>>>(w_out, wo);

    qkv_proj_kernel<<<dim3(3 * E_ / 64, L_ / 256, B_), 256, 0, stream>>>(
        xs, wi, b_in, Qb, Kb, Vb);
    attn_kernel<<<dim3(L_ / 128, H_, B_), 256, 0, stream>>>(Qb, Kb, Vb, attnb);
    out_proj_kernel<<<dim3(E_ / 64, L_ / 256, B_), 256, 0, stream>>>(
        attnb, wo, b_out, out);
}
